// eGIN_module_30021821399387
// MI455X (gfx1250) — matrix operand order repaired and checked by reading, not compiled, not run on hardware
//
#include <hip/hip_runtime.h>

// ---------------------------------------------------------------------------
// Types for CDNA5 WMMA (gfx1250, wave32)
// ---------------------------------------------------------------------------
typedef __bf16 bf16;
typedef bf16  v4bf  __attribute__((ext_vector_type(4)));
typedef bf16  v8bf  __attribute__((ext_vector_type(8)));
typedef bf16  v16bf __attribute__((ext_vector_type(16)));
typedef float v8f   __attribute__((ext_vector_type(8)));

#define BM 128          // C-tile rows per block
#define BN 256          // C-tile cols per block
#define BK 64           // K-slab per barrier (2 WMMA K-steps)
#define NTHREADS 256    // 8 wave32 waves: 2 (M) x 4 (N), each 64x64

// ---------------------------------------------------------------------------
// f32 -> bf16 elementwise convert (vectorized x4)
// ---------------------------------------------------------------------------
__global__ void cvt_bf16_k(const float* __restrict__ in, bf16* __restrict__ out, int n) {
    int i = (blockIdx.x * blockDim.x + threadIdx.x) * 4;
    if (i + 3 < n) {
        float4 v = *(const float4*)(in + i);
        v4bf o;
        o[0] = (bf16)v.x; o[1] = (bf16)v.y; o[2] = (bf16)v.z; o[3] = (bf16)v.w;
        *(v4bf*)(out + i) = o;
    }
}

// ---------------------------------------------------------------------------
// f32 (R x C, row-major) -> bf16 (C x R, row-major) transpose+convert.
// grid = (C/32, R/32, batch), block = (32, 8)
// ---------------------------------------------------------------------------
__global__ void transpose_cvt_k(const float* __restrict__ in, bf16* __restrict__ out,
                                int R, int C) {
    __shared__ float tile[32][33];
    size_t boff = (size_t)blockIdx.z * (size_t)R * (size_t)C;
    in  += boff;
    out += boff;
    int c0 = blockIdx.x * 32;
    int r0 = blockIdx.y * 32;
    int tx = threadIdx.x, ty = threadIdx.y;
#pragma unroll
    for (int i = 0; i < 32; i += 8)
        tile[ty + i][tx] = in[(size_t)(r0 + ty + i) * C + (c0 + tx)];
    __syncthreads();
#pragma unroll
    for (int i = 0; i < 32; i += 8)
        out[(size_t)(c0 + ty + i) * R + (r0 + tx)] = (bf16)tile[tx][ty + i];
}

// ---------------------------------------------------------------------------
// WMMA fragment loaders (LDS -> VGPR), contiguous K order used by both A and B (not AMD's table order):
//   A 16x32: lane (l&15)=row M, half=(l>>4): 16 contiguous K values per lane
//            starting at K=half*16  -> two b128 reads, concatenated.
//   B 32x16: lane (l&15)=col N, half=(l>>4): 16 contiguous K values per lane
//            starting at K=half*16 (N-major storage -> one 32B read).
// ---------------------------------------------------------------------------
__device__ __forceinline__ v16bf frag_a(const bf16* s, int mbase, int ks, int lane) {
    const bf16* p = s + (mbase + (lane & 15)) * BK + ks * 32 + (lane >> 4) * 16;
    v8bf lo = *(const v8bf*)p;
    v8bf hi = *(const v8bf*)(p + 8);
    return __builtin_shufflevector(lo, hi, 0, 1, 2, 3, 4, 5, 6, 7,
                                           8, 9, 10, 11, 12, 13, 14, 15);
}

__device__ __forceinline__ v16bf frag_b(const bf16* s, int nbase, int ks, int lane) {
    const bf16* p = s + (nbase + (lane & 15)) * BK + ks * 32 + (lane >> 4) * 16;
    return *(const v16bf*)p;   // 32B aligned (row stride 128B, offsets 0/32/64/96)
}

// ---------------------------------------------------------------------------
// 128x256 bf16 WMMA GEMM, C = A(MxK) * Bt(NxK)^T, f32 accumulate.
// MODE 0: out_bf16 = acc + (1+eps)*xf          (GIN aggregation)
// MODE 1: out_bf16 = relu(acc + bias[n])
// MODE 2: out_f32  = relu(acc + bias[n])
// blockIdx.z batches A / Bt / xf / out by M*K, N*K, M*N, M*N respectively.
// ---------------------------------------------------------------------------
template <int MODE>
__global__ __launch_bounds__(NTHREADS)
void gemm_wmma_k(const bf16* __restrict__ A, const bf16* __restrict__ Bt,
                 const float* __restrict__ xf, const float* __restrict__ eps,
                 const float* __restrict__ bias,
                 bf16* __restrict__ outb, float* __restrict__ outf,
                 int M, int N, int K) {
    __shared__ __attribute__((aligned(32))) bf16 As[BM * BK];   // 16 KB
    __shared__ __attribute__((aligned(32))) bf16 Bs[BN * BK];   // 32 KB

    const int tid  = threadIdx.x;
    const int lane = tid & 31;
    const int wid  = tid >> 5;      // 8 waves
    const int wm   = wid & 1;       // 2 waves along M (64 rows each)
    const int wn   = wid >> 1;      // 4 waves along N (64 cols each)

    const size_t zA = (size_t)blockIdx.z * (size_t)M * (size_t)K;
    const size_t zB = (size_t)blockIdx.z * (size_t)N * (size_t)K;
    const size_t zC = (size_t)blockIdx.z * (size_t)M * (size_t)N;
    const bf16* Ab = A  + zA;
    const bf16* Bb = Bt + zB;

    const int row0 = blockIdx.y * BM;
    const int col0 = blockIdx.x * BN;

    v8f acc[16];
#pragma unroll
    for (int i = 0; i < 16; ++i)
        acc[i] = (v8f){0.f, 0.f, 0.f, 0.f, 0.f, 0.f, 0.f, 0.f};

    for (int kb = 0; kb < K; kb += BK) {
        // ---- Fill LDS slabs: A 128x64, B 256x64 (bf16), 16B chunks ----
#pragma unroll
        for (int i = 0; i < 4; ++i) {               // A: 1024 chunks / 256 thr
            int c  = tid + i * NTHREADS;
            int r  = c >> 3;                        // 8 chunks per 64-elem row
            int kc = (c & 7) << 3;
            *(v8bf*)(As + r * BK + kc) =
                *(const v8bf*)(Ab + (size_t)(row0 + r) * K + kb + kc);
        }
#pragma unroll
        for (int i = 0; i < 8; ++i) {               // B: 2048 chunks / 256 thr
            int c  = tid + i * NTHREADS;
            int r  = c >> 3;
            int kc = (c & 7) << 3;
            *(v8bf*)(Bs + r * BK + kc) =
                *(const v8bf*)(Bb + (size_t)(col0 + r) * K + kb + kc);
        }
        __syncthreads();

        // ---- Compute: 2 WMMA K-steps x (4x4) tiles = 32 WMMAs per wave ----
#pragma unroll
        for (int ks = 0; ks < 2; ++ks) {
            v16bf a0 = frag_a(As, wm * 64,      ks, lane);
            v16bf a1 = frag_a(As, wm * 64 + 16, ks, lane);
            v16bf a2 = frag_a(As, wm * 64 + 32, ks, lane);
            v16bf a3 = frag_a(As, wm * 64 + 48, ks, lane);
            v16bf b0 = frag_b(Bs, wn * 64,      ks, lane);
            v16bf b1 = frag_b(Bs, wn * 64 + 16, ks, lane);
            v16bf b2 = frag_b(Bs, wn * 64 + 32, ks, lane);
            v16bf b3 = frag_b(Bs, wn * 64 + 48, ks, lane);

            acc[ 0] = __builtin_amdgcn_wmma_f32_16x16x32_bf16(false, a0, false, b0, (short)0, acc[ 0], false, false);
            acc[ 1] = __builtin_amdgcn_wmma_f32_16x16x32_bf16(false, a0, false, b1, (short)0, acc[ 1], false, false);
            acc[ 2] = __builtin_amdgcn_wmma_f32_16x16x32_bf16(false, a0, false, b2, (short)0, acc[ 2], false, false);
            acc[ 3] = __builtin_amdgcn_wmma_f32_16x16x32_bf16(false, a0, false, b3, (short)0, acc[ 3], false, false);
            acc[ 4] = __builtin_amdgcn_wmma_f32_16x16x32_bf16(false, a1, false, b0, (short)0, acc[ 4], false, false);
            acc[ 5] = __builtin_amdgcn_wmma_f32_16x16x32_bf16(false, a1, false, b1, (short)0, acc[ 5], false, false);
            acc[ 6] = __builtin_amdgcn_wmma_f32_16x16x32_bf16(false, a1, false, b2, (short)0, acc[ 6], false, false);
            acc[ 7] = __builtin_amdgcn_wmma_f32_16x16x32_bf16(false, a1, false, b3, (short)0, acc[ 7], false, false);
            acc[ 8] = __builtin_amdgcn_wmma_f32_16x16x32_bf16(false, a2, false, b0, (short)0, acc[ 8], false, false);
            acc[ 9] = __builtin_amdgcn_wmma_f32_16x16x32_bf16(false, a2, false, b1, (short)0, acc[ 9], false, false);
            acc[10] = __builtin_amdgcn_wmma_f32_16x16x32_bf16(false, a2, false, b2, (short)0, acc[10], false, false);
            acc[11] = __builtin_amdgcn_wmma_f32_16x16x32_bf16(false, a2, false, b3, (short)0, acc[11], false, false);
            acc[12] = __builtin_amdgcn_wmma_f32_16x16x32_bf16(false, a3, false, b0, (short)0, acc[12], false, false);
            acc[13] = __builtin_amdgcn_wmma_f32_16x16x32_bf16(false, a3, false, b1, (short)0, acc[13], false, false);
            acc[14] = __builtin_amdgcn_wmma_f32_16x16x32_bf16(false, a3, false, b2, (short)0, acc[14], false, false);
            acc[15] = __builtin_amdgcn_wmma_f32_16x16x32_bf16(false, a3, false, b3, (short)0, acc[15], false, false);
        }
        __syncthreads();
    }

    // Epilogue. C/D layout: VGPR r, lane l -> M = r + 8*(l>>4), N = (l&15).
    const int half = lane >> 4;
    const int nn   = lane & 15;
    const float epsv = (MODE == 0) ? (1.0f + eps[0]) : 0.0f;

#pragma unroll
    for (int tm = 0; tm < 4; ++tm) {
#pragma unroll
        for (int tn = 0; tn < 4; ++tn) {
            v8f c = acc[tm * 4 + tn];
            const int gcol = col0 + wn * 64 + tn * 16 + nn;
            const float bv = (MODE != 0) ? bias[gcol] : 0.0f;
#pragma unroll
            for (int r = 0; r < 8; ++r) {
                const int grow = row0 + wm * 64 + tm * 16 + half * 8 + r;
                const size_t idx = zC + (size_t)grow * N + gcol;
                float v = c[r];
                if (MODE == 0) {
                    v += epsv * xf[idx];
                    outb[idx] = (bf16)v;
                } else {
                    v = fmaxf(v + bv, 0.0f);
                    if (MODE == 2) outf[idx] = v;
                    else           outb[idx] = (bf16)v;
                }
            }
        }
    }
}

// ---------------------------------------------------------------------------
// Launcher.  Inputs: adj, x, eps, W1, b1, W2, b2, W3, b3 (all f32).
// ---------------------------------------------------------------------------
extern "C" void kernel_launch(void* const* d_in, const int* in_sizes, int n_in,
                              void* d_out, int out_size, void* d_ws, size_t ws_size,
                              hipStream_t stream) {
    constexpr int B  = 64,  Nn = 512, F = 512;
    constexpr int H1 = 2048, H2 = 2048, H3 = 512;
    constexpr int M  = B * Nn;                // 32768 flattened rows

    const float* adj = (const float*)d_in[0];
    const float* x   = (const float*)d_in[1];
    const float* eps = (const float*)d_in[2];
    const float* W1  = (const float*)d_in[3];
    const float* b1  = (const float*)d_in[4];
    const float* W2  = (const float*)d_in[5];
    const float* b2  = (const float*)d_in[6];
    const float* W3  = (const float*)d_in[7];
    const float* b3  = (const float*)d_in[8];
    float* out = (float*)d_out;

    // Workspace carve-up (bf16 buffers)
    char* ws = (char*)d_ws;
    bf16* adjb = (bf16*)ws; ws += (size_t)B * Nn * Nn * sizeof(bf16);   // 32 MB
    bf16* xtb  = (bf16*)ws; ws += (size_t)B * F  * Nn * sizeof(bf16);   // 32 MB (per-batch x^T)
    bf16* w1t  = (bf16*)ws; ws += (size_t)H1 * F  * sizeof(bf16);       //  2 MB (N x K)
    bf16* w2t  = (bf16*)ws; ws += (size_t)H2 * H1 * sizeof(bf16);       //  8 MB
    bf16* w3t  = (bf16*)ws; ws += (size_t)H3 * H2 * sizeof(bf16);       //  2 MB
    bf16* h0b  = (bf16*)ws; ws += (size_t)M * F  * sizeof(bf16);        // 32 MB
    bf16* h1b  = (bf16*)ws; ws += (size_t)M * H1 * sizeof(bf16);        // 128 MB
    bf16* h2b  = (bf16*)ws; ws += (size_t)M * H2 * sizeof(bf16);        // 128 MB

    // --- Precision conversion / layout transforms ---
    {
        int n = B * Nn * Nn;
        cvt_bf16_k<<<n / (4 * 256), 256, 0, stream>>>(adj, adjb, n);
    }
    transpose_cvt_k<<<dim3(F / 32, Nn / 32, B), dim3(32, 8), 0, stream>>>(x,  xtb, Nn, F);
    transpose_cvt_k<<<dim3(H1 / 32, F  / 32, 1), dim3(32, 8), 0, stream>>>(W1, w1t, F,  H1);
    transpose_cvt_k<<<dim3(H2 / 32, H1 / 32, 1), dim3(32, 8), 0, stream>>>(W2, w2t, H1, H2);
    transpose_cvt_k<<<dim3(H3 / 32, H2 / 32, 1), dim3(32, 8), 0, stream>>>(W3, w3t, H2, H3);

    // --- GEMM chain (all WMMA bf16, f32 accumulate) ---
    // h0 = (1+eps)*x + adj @ x         (batched 512x512x512)
    gemm_wmma_k<0><<<dim3(F / BN, Nn / BM, B), NTHREADS, 0, stream>>>(
        adjb, xtb, x, eps, nullptr, h0b, nullptr, Nn, F, Nn);
    // h1 = relu(h0 @ W1 + b1)          (32768 x 2048 x 512)
    gemm_wmma_k<1><<<dim3(H1 / BN, M / BM, 1), NTHREADS, 0, stream>>>(
        h0b, w1t, nullptr, nullptr, b1, h1b, nullptr, M, H1, F);
    // h2 = relu(h1 @ W2 + b2)          (32768 x 2048 x 2048)
    gemm_wmma_k<1><<<dim3(H2 / BN, M / BM, 1), NTHREADS, 0, stream>>>(
        h1b, w2t, nullptr, nullptr, b2, h2b, nullptr, M, H2, H1);
    // out = relu(h2 @ W3 + b3)         (32768 x 512 x 2048), f32 output
    gemm_wmma_k<2><<<dim3(H3 / BN, M / BM, 1), NTHREADS, 0, stream>>>(
        h2b, w3t, nullptr, nullptr, b3, nullptr, out, M, H3, H1);
}
